// VectorQuantization_67259187855557
// MI455X (gfx1250) — matrix operand order repaired and checked by reading, compiled, not run on hardware
//
#include <hip/hip_runtime.h>

#define N_TOK   32768
#define K_CODES 8192
#define DIM     256
#define BM      128
#define BN      128
#define THREADS 512
#define LDA     264   // 256 + 8 pad (ushort units), keeps 16B alignment, breaks bank conflicts

typedef __attribute__((ext_vector_type(16))) __bf16          v16bf;
typedef __attribute__((ext_vector_type(8)))  float           v8f;
typedef __attribute__((ext_vector_type(16))) unsigned short  v16us;
typedef __attribute__((ext_vector_type(8)))  unsigned short  v8us;

union Frag16 {
  v16bf bf;
  v16us us;
  v8us  h[2];
};

__device__ __forceinline__ unsigned short bf16_rne(float f) {
  unsigned u = __float_as_uint(f);
  u += 0x7FFFu + ((u >> 16) & 1u);           // round-to-nearest-even
  return (unsigned short)(u >> 16);
}

__device__ __forceinline__ void bf16_split(float f, unsigned short& hi, unsigned short& lo) {
  unsigned short h = bf16_rne(f);
  float fh = __uint_as_float(((unsigned)h) << 16);
  hi = h;
  lo = bf16_rne(f - fh);                      // residual, ~2^-9 smaller
}

__device__ __forceinline__ v8f wmma_bf16(v16bf a, v16bf b, v8f c) {
  // (neg_a, A, neg_b, B, c_mod, C, reuse_a, reuse_b) -> v_wmma_f32_16x16x32_bf16
  return __builtin_amdgcn_wmma_f32_16x16x32_bf16(false, a, false, b, (short)0, c, false, false);
}

// ---- prep 1: c2[k] = ||E_k||^2 ----
__global__ void vq_c2_kernel(const float* __restrict__ cb, float* __restrict__ c2) {
  int k = blockIdx.x * blockDim.x + threadIdx.x;
  if (k >= K_CODES) return;
  const float4* p = (const float4*)(cb + (size_t)k * DIM);
  float s = 0.f;
#pragma unroll 8
  for (int i = 0; i < DIM / 4; ++i) {
    float4 v = p[i];
    s += v.x * v.x + v.y * v.y + v.z * v.z + v.w * v.w;
  }
  c2[k] = s;
}

// ---- prep 2: one-time codebook fp32 -> bf16 hi/lo split (kills inner-loop VALU) ----
__global__ void vq_cvt_kernel(const float* __restrict__ cb,
                              unsigned short* __restrict__ cbh,
                              unsigned short* __restrict__ cbl) {
  size_t i = (size_t)(blockIdx.x * blockDim.x + threadIdx.x);   // one float4 per thread
  if (i >= (size_t)K_CODES * (DIM / 4)) return;
  float4 v = ((const float4*)cb)[i];
  unsigned short h0, h1, h2, h3, l0, l1, l2, l3;
  bf16_split(v.x, h0, l0);
  bf16_split(v.y, h1, l1);
  bf16_split(v.z, h2, l2);
  bf16_split(v.w, h3, l3);
  uint2 hp, lp;
  hp.x = (unsigned)h0 | ((unsigned)h1 << 16);
  hp.y = (unsigned)h2 | ((unsigned)h3 << 16);
  lp.x = (unsigned)l0 | ((unsigned)l1 << 16);
  lp.y = (unsigned)l2 | ((unsigned)l3 << 16);
  ((uint2*)cbh)[i] = hp;
  ((uint2*)cbl)[i] = lp;
}

// ---- main: fused GEMM (bf16x3 split) + running argmin + gather ----
// PRECONV=true : B frags loaded as pre-split bf16 from workspace (hot loop = loads + wmma only)
// PRECONV=false: fallback, on-the-fly split (used only if workspace is too small)
template <bool PRECONV>
__launch_bounds__(THREADS)
__global__ void vq_main_kernel(const float* __restrict__ z,
                               const float* __restrict__ cb,
                               const unsigned short* __restrict__ cbh,
                               const unsigned short* __restrict__ cbl,
                               const float* __restrict__ c2,
                               float* __restrict__ zq,
                               float* __restrict__ out_idx) {
  extern __shared__ unsigned short smem[];         // 2 * BM * LDA ushorts = 135168 B
  unsigned short* Ahi = smem;
  unsigned short* Alo = smem + BM * LDA;
  __shared__ unsigned bestIdxSh[BM];

  const int tid = threadIdx.x;
  const int rowBase = blockIdx.x * BM;

  // Stage z tile into LDS as bf16 hi/lo (converted once per block, amortized over 64 k-tiles)
  for (int i = tid; i < BM * (DIM / 4); i += THREADS) {
    int r  = i >> 6;          // DIM/4 == 64
    int c4 = i & 63;
    float4 v = ((const float4*)(z + (size_t)(rowBase + r) * DIM))[c4];
    int base = r * LDA + c4 * 4;
    bf16_split(v.x, Ahi[base + 0], Alo[base + 0]);
    bf16_split(v.y, Ahi[base + 1], Alo[base + 1]);
    bf16_split(v.z, Ahi[base + 2], Alo[base + 2]);
    bf16_split(v.w, Ahi[base + 3], Alo[base + 3]);
  }
  __syncthreads();

  const int wave = tid >> 5;
  const int lane = tid & 31;
  const int mw   = wave >> 2;     // wave row  (0..3)
  const int nw   = wave & 3;      // wave col  (0..3)
  const int half = lane >> 4;     // lane half (0/1)
  const int l16  = lane & 15;

  float    bestS[2][8];
  unsigned bestK[2][8];
#pragma unroll
  for (int mf = 0; mf < 2; ++mf)
#pragma unroll
    for (int v = 0; v < 8; ++v) { bestS[mf][v] = 3.4e38f; bestK[mf][v] = 0u; }

  const int rowA0 = mw * 32 + l16;        // A frag row, mf = 0
  const int rowA1 = rowA0 + 16;           // mf = 1

  for (int kt = 0; kt < K_CODES / BN; ++kt) {
    const int kc0 = kt * BN + nw * 32 + l16;   // codebook row for nf=0 column
    const int kc1 = kc0 + 16;                  // nf=1
    v8f acc00 = {}, acc01 = {}, acc10 = {}, acc11 = {};

#pragma unroll 2
    for (int d0 = 0; d0 < DIM; d0 += 32) {
      // ---- A fragments (16x32 bf16 layout: lane<16 -> K 0..15, lane>=16 -> K 16..31) ----
      Frag16 ah0, al0, ah1, al1;
      {
        int a = rowA0 * LDA + d0 + half * 16;
        ah0.h[0] = *(const v8us*)(Ahi + a);
        ah0.h[1] = *(const v8us*)(Ahi + a + 8);
        al0.h[0] = *(const v8us*)(Alo + a);
        al0.h[1] = *(const v8us*)(Alo + a + 8);
        a = rowA1 * LDA + d0 + half * 16;
        ah1.h[0] = *(const v8us*)(Ahi + a);
        ah1.h[1] = *(const v8us*)(Ahi + a + 8);
        al1.h[0] = *(const v8us*)(Alo + a);
        al1.h[1] = *(const v8us*)(Alo + a + 8);
      }
      // ---- B fragments (32x16 bf16 layout: lane<16 -> K 0..15, lane>=16 -> K 16..31, N = lane&15) ----
      Frag16 bh0, bl0, bh1, bl1;
      if constexpr (PRECONV) {
        // Pre-split codebook: pure b128 loads, no conversion VALU in the hot loop
        size_t o0 = (size_t)kc0 * DIM + d0 + half * 16;
        size_t o1 = (size_t)kc1 * DIM + d0 + half * 16;
        bh0.h[0] = *(const v8us*)(cbh + o0);
        bh0.h[1] = *(const v8us*)(cbh + o0 + 8);
        bl0.h[0] = *(const v8us*)(cbl + o0);
        bl0.h[1] = *(const v8us*)(cbl + o0 + 8);
        bh1.h[0] = *(const v8us*)(cbh + o1);
        bh1.h[1] = *(const v8us*)(cbh + o1 + 8);
        bl1.h[0] = *(const v8us*)(cbl + o1);
        bl1.h[1] = *(const v8us*)(cbl + o1 + 8);
      } else {
        const float* p0 = cb + (size_t)kc0 * DIM + d0 + half * 16;
        float b[16];
        *(float4*)(b + 0)  = ((const float4*)p0)[0];
        *(float4*)(b + 4)  = ((const float4*)p0)[1];
        *(float4*)(b + 8)  = ((const float4*)p0)[2];
        *(float4*)(b + 12) = ((const float4*)p0)[3];
#pragma unroll
        for (int i = 0; i < 16; ++i) {
          unsigned short h, l; bf16_split(b[i], h, l);
          bh0.us[i] = h; bl0.us[i] = l;
        }
        const float* p1 = cb + (size_t)kc1 * DIM + d0 + half * 16;
        *(float4*)(b + 0)  = ((const float4*)p1)[0];
        *(float4*)(b + 4)  = ((const float4*)p1)[1];
        *(float4*)(b + 8)  = ((const float4*)p1)[2];
        *(float4*)(b + 12) = ((const float4*)p1)[3];
#pragma unroll
        for (int i = 0; i < 16; ++i) {
          unsigned short h, l; bf16_split(b[i], h, l);
          bh1.us[i] = h; bl1.us[i] = l;
        }
      }
      // ---- bf16x3 product per (mf,nf): hh + hl + lh ~= fp32-accurate dot ----
      acc00 = wmma_bf16(ah0.bf, bh0.bf, acc00);
      acc00 = wmma_bf16(ah0.bf, bl0.bf, acc00);
      acc00 = wmma_bf16(al0.bf, bh0.bf, acc00);

      acc01 = wmma_bf16(ah0.bf, bh1.bf, acc01);
      acc01 = wmma_bf16(ah0.bf, bl1.bf, acc01);
      acc01 = wmma_bf16(al0.bf, bh1.bf, acc01);

      acc10 = wmma_bf16(ah1.bf, bh0.bf, acc10);
      acc10 = wmma_bf16(ah1.bf, bl0.bf, acc10);
      acc10 = wmma_bf16(al1.bf, bh0.bf, acc10);

      acc11 = wmma_bf16(ah1.bf, bh1.bf, acc11);
      acc11 = wmma_bf16(ah1.bf, bl1.bf, acc11);
      acc11 = wmma_bf16(al1.bf, bh1.bf, acc11);
    }

    // ---- fused epilogue: score = ||E||^2 - 2*dot, running per-lane argmin ----
    const float c20 = c2[kc0];
    const float c21 = c2[kc1];
#pragma unroll
    for (int v = 0; v < 8; ++v) {
      float s;
      s = fmaf(-2.f, acc00[v], c20); if (s < bestS[0][v]) { bestS[0][v] = s; bestK[0][v] = (unsigned)kc0; }
      s = fmaf(-2.f, acc01[v], c21); if (s < bestS[0][v]) { bestS[0][v] = s; bestK[0][v] = (unsigned)kc1; }
      s = fmaf(-2.f, acc10[v], c20); if (s < bestS[1][v]) { bestS[1][v] = s; bestK[1][v] = (unsigned)kc0; }
      s = fmaf(-2.f, acc11[v], c21); if (s < bestS[1][v]) { bestS[1][v] = s; bestK[1][v] = (unsigned)kc1; }
    }
  }

  // ---- cross-lane / cross-wave argmin reduction via LDS (reuse A region) ----
  __syncthreads();
  float*    reds = (float*)smem;                 // [BM][64] scores  (32 KB)
  unsigned* redi = ((unsigned*)smem) + BM * 64;  // [BM][64] indices (32 KB)

#pragma unroll
  for (int mf = 0; mf < 2; ++mf) {
#pragma unroll
    for (int v = 0; v < 8; ++v) {
      int row  = mw * 32 + mf * 16 + half * 8 + v;   // C-matrix row mapping
      int slot = nw * 16 + l16;                      // 4 waves x 16 columns
      reds[row * 64 + slot] = bestS[mf][v];
      redi[row * 64 + slot] = bestK[mf][v];
    }
  }
  __syncthreads();

  if (tid < BM) {
    float    bs = reds[tid * 64];
    unsigned bk = redi[tid * 64];
    for (int s = 1; s < 64; ++s) {
      float    v  = reds[tid * 64 + s];
      unsigned k2 = redi[tid * 64 + s];
      if (v < bs || (v == bs && k2 < bk)) { bs = v; bk = k2; }   // argmin, first-index tie-break
    }
    bestIdxSh[tid] = bk;
    out_idx[rowBase + tid] = (float)bk;   // indices emitted as float values
  }
  __syncthreads();

  // ---- gather z_q rows from codebook (4 threads per row) ----
  {
    int r    = tid >> 2;
    int part = tid & 3;
    unsigned bi = bestIdxSh[r];
    const float4* src = (const float4*)(cb + (size_t)bi * DIM + part * 64);
    float4*       dst = (float4*)(zq + (size_t)(rowBase + r) * DIM + part * 64);
#pragma unroll
    for (int i = 0; i < 16; ++i) dst[i] = src[i];
  }
}

extern "C" void kernel_launch(void* const* d_in, const int* in_sizes, int n_in,
                              void* d_out, int out_size, void* d_ws, size_t ws_size,
                              hipStream_t stream) {
  const float* z  = (const float*)d_in[0];   // [8,4096,256] f32
  const float* cb = (const float*)d_in[1];   // [8192,256]   f32
  float* out     = (float*)d_out;
  float* zq      = out;                                  // N*D floats
  float* out_idx = out + (size_t)N_TOK * DIM;            // N floats (indices as float)
  float* c2      = (float*)d_ws;                         // 8192 f32 = 32 KB scratch

  const size_t c2_bytes  = (size_t)K_CODES * sizeof(float);                 // 32 KB
  const size_t cvt_bytes = (size_t)K_CODES * DIM * sizeof(unsigned short);  // 4 MB each
  const size_t shmem     = (size_t)2 * BM * LDA * sizeof(unsigned short);   // 135168 B LDS

  vq_c2_kernel<<<K_CODES / 256, 256, 0, stream>>>(cb, c2);

  if (ws_size >= c2_bytes + 2 * cvt_bytes) {
    // Fast path: one-time codebook bf16 hi/lo split into workspace
    unsigned short* cbh = (unsigned short*)((char*)d_ws + c2_bytes);
    unsigned short* cbl = cbh + (size_t)K_CODES * DIM;
    const int n4 = K_CODES * (DIM / 4);
    vq_cvt_kernel<<<(n4 + 255) / 256, 256, 0, stream>>>(cb, cbh, cbl);
    vq_main_kernel<true><<<N_TOK / BM, THREADS, shmem, stream>>>(z, cb, cbh, cbl, c2, zq, out_idx);
  } else {
    // Fallback: on-the-fly split (correct, more VALU in hot loop)
    vq_main_kernel<false><<<N_TOK / BM, THREADS, shmem, stream>>>(z, cb, nullptr, nullptr, c2, zq, out_idx);
  }
}
